// HeteroMLPPredictor_89249420411231
// MI455X (gfx1250) — matrix operand order repaired and checked by reading, not compiled, not run on hardware
//
#include <hip/hip_runtime.h>
#include <hip/hip_bf16.h>
#include <cstdint>
#include <cstddef>

// ---------------------------------------------------------------------------
// HeteroMLPPredictor fused pipeline for gfx1250 (MI455X, wave32, WMMA).
//   pack_w_kernel   : f32 weights -> f16 WMMA B-fragment layout
//   node_linear_f16 : u_f = h@W+b  (f16 out, row-major), persistent waves
//   edge_mlp        : gather+add+leaky -> WMMA vs W1 -> leaky -> dot W2 -> leaky
// Design notes (roofline): edge phase reads 512 B/edge of f16 features; both
// feature tables (38.4 MB) fit in the 192 MB L2, so the phase is L2-BW bound.
// Persistent waves amortize the 32 KB LDS weight fill; all K-tile global
// loads are issued before any consumption to maximize loads-in-flight.
// ---------------------------------------------------------------------------

typedef __attribute__((ext_vector_type(16))) _Float16 v16h;
typedef __attribute__((ext_vector_type(8)))  float    v8f;

#define DMODEL 128
#define KT 4                 // 128 / 32 (K per wmma)
#define NT 8                 // 128 / 16 (N per wmma)
#define NFRAG (KT*NT)        // 32 fragments per weight matrix
#define FRAG_U4 64           // uint4 per fragment (32 lanes * 32B)
#define PACK_HALF (NFRAG*512)// 16384 f16 per packed matrix (32 KB)

union H8  { uint4 q;    _Float16 h[8]; };
union V16 { v16h  v;    uint4 q[2];    };

__device__ __forceinline__ float leaky(float x) {
    return x >= 0.0f ? x : 0.01f * x;
}

// ---------------------------------------------------------------------------
// Pack W (f32, row-major [K][N], K=N=128) into f16 WMMA B fragments.
// Fragment f = kt*8+nt ; lane l : N = nt*16 + (l&15), K = kt*32 + (l>>4)*8 + (e&7) + (e>>3)*16
// stored as P[f*512 + l*16 + e].
// ---------------------------------------------------------------------------
__global__ void __launch_bounds__(256)
pack_w_kernel(const float* __restrict__ W0, const float* __restrict__ W1s,
              const float* __restrict__ W2s,
              _Float16* __restrict__ P0, _Float16* __restrict__ P1,
              _Float16* __restrict__ P2) {
    int idx = blockIdx.x * blockDim.x + threadIdx.x;
    if (idx >= 3 * PACK_HALF) return;
    int mat = idx / PACK_HALF;
    int rem = idx - mat * PACK_HALF;
    int f = rem >> 9;            // fragment
    int l = (rem >> 4) & 31;     // lane
    int e = rem & 15;            // element within lane
    int kt = f >> 3, nt = f & 7;
    int n = nt * 16 + (l & 15);
    int k = kt * 32 + (l >> 4) * 8 + (e & 7) + (e >> 3) * 16;
    const float* W = (mat == 0) ? W0 : (mat == 1) ? W1s : W2s;
    _Float16*    P = (mat == 0) ? P0 : (mat == 1) ? P1 : P2;
    P[rem] = (_Float16)W[k * DMODEL + n];
}

// ---------------------------------------------------------------------------
// out[f16, row-major] = (h @ W + b); persistent waves, 16-row tile / iter.
// ---------------------------------------------------------------------------
__global__ void __launch_bounds__(256)
node_linear_f16(const float* __restrict__ h,
                const _Float16* __restrict__ Wpk,
                const float* __restrict__ bias,
                _Float16* __restrict__ out,
                int nrows) {
    __shared__ uint4 ldsW[NFRAG * FRAG_U4];   // 32 KB packed weight
    const int tid = threadIdx.x;
    {
        const uint4* g = (const uint4*)Wpk;
        #pragma unroll
        for (int j = 0; j < 8; ++j) ldsW[tid + 256 * j] = g[tid + 256 * j];
    }
    __syncthreads();

    const int wave = tid >> 5;
    const int lane = tid & 31;
    const int nl   = lane & 15;
    const int hf   = lane >> 4;
    const int h8   = hf * 8;

    const int ntiles = (nrows + 15) >> 4;
    const int wstride = gridDim.x * 8;

    float bv[NT];
    #pragma unroll
    for (int nt = 0; nt < NT; ++nt) bv[nt] = bias[nt * 16 + nl];

    for (int tile = blockIdx.x * 8 + wave; tile < ntiles; tile += wstride) {
        int row = tile * 16 + nl;
        if (row >= nrows) row = nrows - 1;      // clamp; stores guarded below
        const float* hrow = h + (size_t)row * DMODEL;

        // issue all global loads for all K-tiles first (max loads in flight)
        float4 raw[KT][4];
        #pragma unroll
        for (int kt = 0; kt < KT; ++kt) {
            const float* hp = hrow + kt * 32 + h8;
            raw[kt][0] = *(const float4*)(hp);
            raw[kt][1] = *(const float4*)(hp + 4);
            raw[kt][2] = *(const float4*)(hp + 16);
            raw[kt][3] = *(const float4*)(hp + 20);
        }

        v16h afr[KT];
        #pragma unroll
        for (int kt = 0; kt < KT; ++kt) {
            afr[kt][0]=(_Float16)raw[kt][0].x;  afr[kt][1]=(_Float16)raw[kt][0].y;
            afr[kt][2]=(_Float16)raw[kt][0].z;  afr[kt][3]=(_Float16)raw[kt][0].w;
            afr[kt][4]=(_Float16)raw[kt][1].x;  afr[kt][5]=(_Float16)raw[kt][1].y;
            afr[kt][6]=(_Float16)raw[kt][1].z;  afr[kt][7]=(_Float16)raw[kt][1].w;
            afr[kt][8]=(_Float16)raw[kt][2].x;  afr[kt][9]=(_Float16)raw[kt][2].y;
            afr[kt][10]=(_Float16)raw[kt][2].z; afr[kt][11]=(_Float16)raw[kt][2].w;
            afr[kt][12]=(_Float16)raw[kt][3].x; afr[kt][13]=(_Float16)raw[kt][3].y;
            afr[kt][14]=(_Float16)raw[kt][3].z; afr[kt][15]=(_Float16)raw[kt][3].w;
        }

        v8f acc[NT];
        #pragma unroll
        for (int nt = 0; nt < NT; ++nt) acc[nt] = {};

        #pragma unroll
        for (int kt = 0; kt < KT; ++kt) {
            #pragma unroll
            for (int nt = 0; nt < NT; ++nt) {
                const int f = kt * NT + nt;
                V16 bb;
                bb.q[0] = ldsW[f * FRAG_U4 + lane * 2];
                bb.q[1] = ldsW[f * FRAG_U4 + lane * 2 + 1];
                acc[nt] = __builtin_amdgcn_wmma_f32_16x16x32_f16(
                    false, afr[kt], false, bb.v, (short)0, acc[nt], false, false);
            }
        }

        #pragma unroll
        for (int nt = 0; nt < NT; ++nt) {
            #pragma unroll
            for (int r = 0; r < 8; ++r) {
                int rowOut = tile * 16 + hf * 8 + r;   // C: M = r + 8*(lane>>4)
                if (rowOut < nrows)
                    out[(size_t)rowOut * DMODEL + nt * 16 + nl] =
                        (_Float16)(acc[nt][r] + bv[nt]);
            }
        }
    }
}

// ---------------------------------------------------------------------------
// Per 16-edge tile: gather u_f[src]+i_f[dst] (f16), leaky, WMMA vs W1,
// bias+leaky, dot W2 + b2, leaky, store f32. Persistent waves.
// ---------------------------------------------------------------------------
__global__ void __launch_bounds__(256)
edge_mlp(const _Float16* __restrict__ uf,
         const _Float16* __restrict__ vfm,
         const int* __restrict__ src,
         const int* __restrict__ dst,
         const _Float16* __restrict__ W1pk,
         const float* __restrict__ b1,
         const float* __restrict__ W2,
         const float* __restrict__ b2,
         float* __restrict__ out,
         int E) {
    __shared__ uint4 ldsW[NFRAG * FRAG_U4];   // 32 KB packed W1
    const int tid = threadIdx.x;
    {
        const uint4* g = (const uint4*)W1pk;
        #pragma unroll
        for (int j = 0; j < 8; ++j) ldsW[tid + 256 * j] = g[tid + 256 * j];
    }
    __syncthreads();

    const int wave = tid >> 5;
    const int lane = tid & 31;
    const int nl   = lane & 15;
    const int hf   = lane >> 4;
    const int h8   = hf * 8;

    const int ntiles = (E + 15) >> 4;
    const int wstride = gridDim.x * 8;

    float b1v[NT], w2v[NT];
    #pragma unroll
    for (int nt = 0; nt < NT; ++nt) {
        b1v[nt] = b1[nt * 16 + nl];
        w2v[nt] = W2[nt * 16 + nl];
    }
    const float b2s = b2[0];

    for (int tile = blockIdx.x * 8 + wave; tile < ntiles; tile += wstride) {
        int e = tile * 16 + nl;
        if (e >= E) e = E - 1;                  // clamp; stores guarded below
        const int s = src[e];
        const int d = dst[e];
        const _Float16* ur = uf  + (size_t)s * DMODEL;
        const _Float16* vr = vfm + (size_t)d * DMODEL;

        // issue all 16 gather loads before any consumption
        H8 uraw[KT][2], vraw[KT][2];
        #pragma unroll
        for (int kt = 0; kt < KT; ++kt) {
            const int o0 = kt * 32 + h8;
            uraw[kt][0].q = *(const uint4*)(ur + o0);
            uraw[kt][1].q = *(const uint4*)(ur + o0 + 16);
            vraw[kt][0].q = *(const uint4*)(vr + o0);
            vraw[kt][1].q = *(const uint4*)(vr + o0 + 16);
        }

        v16h afr[KT];
        #pragma unroll
        for (int kt = 0; kt < KT; ++kt) {
            #pragma unroll
            for (int j = 0; j < 8; ++j) {
                float x0 = (float)uraw[kt][0].h[j] + (float)vraw[kt][0].h[j];
                float x1 = (float)uraw[kt][1].h[j] + (float)vraw[kt][1].h[j];
                afr[kt][j]     = (_Float16)leaky(x0);   // score -> leaky
                afr[kt][j + 8] = (_Float16)leaky(x1);
            }
        }

        v8f acc[NT];
        #pragma unroll
        for (int nt = 0; nt < NT; ++nt) acc[nt] = {};

        #pragma unroll
        for (int kt = 0; kt < KT; ++kt) {
            #pragma unroll
            for (int nt = 0; nt < NT; ++nt) {
                const int f = kt * NT + nt;
                V16 bb;
                bb.q[0] = ldsW[f * FRAG_U4 + lane * 2];
                bb.q[1] = ldsW[f * FRAG_U4 + lane * 2 + 1];
                acc[nt] = __builtin_amdgcn_wmma_f32_16x16x32_f16(
                    false, afr[kt], false, bb.v, (short)0, acc[nt], false, false);
            }
        }

        // y[m] = leaky( sum_n leaky(acc[m][n] + b1[n]) * W2[n] + b2 )
        #pragma unroll
        for (int r = 0; r < 8; ++r) {
            float p = 0.0f;
            #pragma unroll
            for (int nt = 0; nt < NT; ++nt)
                p += leaky(acc[nt][r] + b1v[nt]) * w2v[nt];
            // reduce over the 16 lanes sharing M (xor<16 keeps halves apart)
            p += __shfl_xor(p, 1, 32);
            p += __shfl_xor(p, 2, 32);
            p += __shfl_xor(p, 4, 32);
            p += __shfl_xor(p, 8, 32);
            if (nl == 0) {
                int eo = tile * 16 + hf * 8 + r;
                if (eo < E) out[eo] = leaky(p + b2s);
            }
        }
    }
}

// ---------------------------------------------------------------------------
extern "C" void kernel_launch(void* const* d_in, const int* in_sizes, int n_in,
                              void* d_out, int out_size, void* d_ws, size_t ws_size,
                              hipStream_t stream) {
    const float* h_user  = (const float*)d_in[0];
    const float* h_item  = (const float*)d_in[1];
    const int*   src_idx = (const int*)d_in[2];
    const int*   dst_idx = (const int*)d_in[3];
    const float* W_left  = (const float*)d_in[4];
    const float* b_left  = (const float*)d_in[5];
    const float* W_right = (const float*)d_in[6];
    const float* b_right = (const float*)d_in[7];
    const float* W1      = (const float*)d_in[8];
    const float* b1      = (const float*)d_in[9];
    const float* W2      = (const float*)d_in[10];
    const float* b2      = (const float*)d_in[11];
    float* out = (float*)d_out;

    const int NU = in_sizes[0] / DMODEL;
    const int NI = in_sizes[1] / DMODEL;
    const int E  = in_sizes[2];

    // workspace layout (all 16B-aligned)
    char* ws = (char*)d_ws;
    _Float16* uf = (_Float16*)ws;                               // NU*128 f16
    _Float16* vf = uf + (size_t)NU * DMODEL;                    // NI*128 f16
    _Float16* pL = vf + (size_t)NI * DMODEL;                    // packed W_left
    _Float16* pR = pL + PACK_HALF;                              // packed W_right
    _Float16* p1 = pR + PACK_HALF;                              // packed W1

    // 1) pack weights into WMMA fragment layout (f16)
    {
        int total = 3 * PACK_HALF;
        pack_w_kernel<<<(total + 255) / 256, 256, 0, stream>>>(
            W_left, W_right, W1, pL, pR, p1);
    }
    // 2) node linear transforms (WMMA), f16 outputs; persistent waves
    {
        int ntiles = (NU + 15) / 16;
        int blocks = (ntiles + 7) / 8; if (blocks > 512) blocks = 512;
        node_linear_f16<<<blocks, 256, 0, stream>>>(h_user, pL, b_left, uf, NU);
    }
    {
        int ntiles = (NI + 15) / 16;
        int blocks = (ntiles + 7) / 8; if (blocks > 512) blocks = 512;
        node_linear_f16<<<blocks, 256, 0, stream>>>(h_item, pR, b_right, vf, NI);
    }
    // 3) fused edge gather + MLP (WMMA); persistent waves (~8 tiles/wave)
    {
        int ntiles = (E + 15) / 16;
        int blocks = (ntiles + 7) / 8; if (blocks > 1024) blocks = 1024;
        edge_mlp<<<blocks, 256, 0, stream>>>(
            uf, vf, src_idx, dst_idx, p1, b1, W2, b2, out, E);
    }
}
